// AttentionWithDINOKV_11785390260605
// MI455X (gfx1250) — matrix operand order repaired and checked by reading, not compiled, not run on hardware
//
#include <hip/hip_runtime.h>
#include <hip/hip_bf16.h>
#include <math.h>

// Problem constants
#define NUM_HEADS 16
#define HEAD_DIM  72
#define HD_PAD    96          // 72 padded to multiple of 32 for WMMA K loop
#define SEQ       1024
#define BATCH     8
#define CH        1152
#define NQKV      3456
#define NQKV_PAD  3584        // 14 * 256
#define NPROJ_PAD 1280        // 5 * 256
#define BHN       (BATCH * NUM_HEADS)          // 128 head-batches
#define SCALE_F   0.11785113019775793f          // 72^-0.5
#define EPS_F     1e-12f

typedef __attribute__((ext_vector_type(16))) __bf16 v16bf;
typedef __attribute__((ext_vector_type(8)))  float  v8f;

union FragBF {
  v16bf v;
  uint4 q[2];
  unsigned short s[16];
};

// ---------- scalar bf16 helpers (RNE) ----------
__device__ __forceinline__ unsigned short f2bf(float f) {
  unsigned int u = __float_as_uint(f);
  unsigned int r = (u + 0x7FFFu + ((u >> 16) & 1u)) >> 16;
  return (unsigned short)r;
}
__device__ __forceinline__ float bf2f(unsigned short h) {
  return __uint_as_float(((unsigned int)h) << 16);
}

// ---------- WMMA fragment builders (K-contiguous per lane half, same k order for A and B) ----------
__device__ __forceinline__ v16bf frag_from_2(const unsigned short* p0, const unsigned short* p1) {
  FragBF f;
  f.q[0] = *(const uint4*)p0;
  f.q[1] = *(const uint4*)p1;
  return f.v;
}
// 16-bit A 16x32: lane holds row (lane&15); K contiguous [kk+16*hi..+15]
__device__ __forceinline__ v16bf fragA_bf(const unsigned short* base, int stride, int kk, unsigned lane) {
  const unsigned short* row = base + (lane & 15u) * stride + kk + 16 * (lane >> 4);
  return frag_from_2(row, row + 8);
}
// A-fragment from fp32 LDS strip (softmax probabilities), converted to bf16
__device__ __forceinline__ v16bf fragA_f32(const float* base, int stride, int kk, unsigned lane) {
  const float* row = base + (lane & 15u) * stride + kk + 16 * (lane >> 4);
  FragBF f;
#pragma unroll
  for (int e = 0; e < 8; ++e) f.s[e] = f2bf(row[e]);
#pragma unroll
  for (int e = 0; e < 8; ++e) f.s[8 + e] = f2bf(row[8 + e]);
  return f.v;
}
// A-fragment = dino softmax probabilities reconstructed on the fly from raw bf16 logits
__device__ __forceinline__ v16bf fragA_dinoP(const unsigned short* base, int stride, int kk,
                                             unsigned lane, float mx, float inv_es) {
  const unsigned short* row = base + (lane & 15u) * stride + kk + 16 * (lane >> 4);
  FragBF f;
#pragma unroll
  for (int e = 0; e < 8; ++e) f.s[e] = f2bf(__expf(bf2f(row[e]) - mx) * inv_es);
#pragma unroll
  for (int e = 0; e < 8; ++e) f.s[8 + e] = f2bf(__expf(bf2f(row[8 + e]) - mx) * inv_es);
  return f.v;
}

#define WMMA_BF16(A, B, C) \
  __builtin_amdgcn_wmma_f32_16x16x32_bf16(false, (A), false, (B), (short)0, (C), false, false)

// ---------- cooperative global->LDS panel copy (async path if toolchain has it) ----------
#if __has_builtin(__builtin_amdgcn_global_load_async_to_lds_b128) && \
    __has_builtin(__builtin_amdgcn_s_wait_asynccnt)
#define HAVE_ASYNC_LDS 1
// Signature (from hipcc diagnostic): param0 = int __attribute__((vector_size(16))) __device__*
typedef int v4i_vs __attribute__((vector_size(16)));
typedef __attribute__((address_space(1))) v4i_vs* g_v4i_t;
typedef __attribute__((address_space(3))) v4i_vs* l_v4i_t;
#else
#define HAVE_ASYNC_LDS 0
#endif

__device__ __forceinline__ void copy_panel_to_lds(unsigned short* lds_dst,
                                                  const unsigned short* gsrc,
                                                  int n_uint4, int tid, int nthreads) {
#if HAVE_ASYNC_LDS
  for (int i = tid; i < n_uint4; i += nthreads) {
    __builtin_amdgcn_global_load_async_to_lds_b128(
        (g_v4i_t)(gsrc + (size_t)i * 8),
        (l_v4i_t)(lds_dst + (size_t)i * 8), 0, 0);
  }
  __builtin_amdgcn_s_wait_asynccnt(0);
#else
  for (int i = tid; i < n_uint4; i += nthreads)
    ((uint4*)lds_dst)[i] = ((const uint4*)gsrc)[i];
#endif
}

// ==================================================================================
// Pack / init kernels
// ==================================================================================
__global__ void zero_u32_kernel(unsigned int* __restrict__ p, long long n) {
  long long i = (long long)blockIdx.x * blockDim.x + threadIdx.x;
  if (i < n) p[i] = 0u;
}
__global__ void f2bf_kernel(const float* __restrict__ in, unsigned short* __restrict__ out, int n) {
  int i = blockIdx.x * blockDim.x + threadIdx.x;
  if (i < n) out[i] = f2bf(in[i]);
}
// in [rows][cols] fp32 -> out [colsPad][rows] bf16 (zero rows for c >= cols)
__global__ void transpose_pad_bf_kernel(const float* __restrict__ in,
                                        unsigned short* __restrict__ out,
                                        int rows, int cols, int colsPad) {
  int i = blockIdx.x * blockDim.x + threadIdx.x;
  if (i < colsPad * rows) {
    int c = i / rows;
    int r = i - c * rows;
    out[i] = (c < cols) ? f2bf(in[(size_t)r * cols + c]) : (unsigned short)0;
  }
}
// in [nrows][72] fp32 -> out [nrows][96] bf16 (zero-padded tail)
__global__ void pad_bf_kernel(const float* __restrict__ in, unsigned short* __restrict__ out, int nrows) {
  int i = blockIdx.x * blockDim.x + threadIdx.x;
  if (i < nrows * HD_PAD) {
    int row = i / HD_PAD;
    int d = i - row * HD_PAD;
    out[i] = (d < HEAD_DIM) ? f2bf(in[(size_t)row * HEAD_DIM + d]) : (unsigned short)0;
  }
}
// in [BH][1024][72] fp32 -> out [BH][96][1024] bf16 (transposed within head-batch)
__global__ void padT_bf_kernel(const float* __restrict__ in, unsigned short* __restrict__ out) {
  int i = blockIdx.x * blockDim.x + threadIdx.x;
  const int per = HD_PAD * SEQ;
  if (i < BHN * per) {
    int bh = i / per;
    int rem = i - bh * per;
    int d = rem / SEQ;
    int n = rem - d * SEQ;
    out[i] = (d < HEAD_DIM) ? f2bf(in[((size_t)bh * SEQ + n) * HEAD_DIM + d]) : (unsigned short)0;
  }
}

// ==================================================================================
// GEMM core: block = 256 thr (8 waves), tile M=32 x N=256; wave owns 32x32 (2x2 accs).
// A panel [32][1152] staged ONCE in LDS; K-loop is barrier-free (4 wmma / k-step / wave).
// ==================================================================================
__device__ __forceinline__ void gemm_core_32x32(const unsigned short* Apan,  // LDS [32][CH]
                                                const unsigned short* wT,    // [Npad][CH]
                                                int colw,                    // wave col base
                                                unsigned lane, v8f acc[4]) {
  const unsigned hi = lane >> 4;
  const unsigned short* b0 = wT + (size_t)(colw + (int)(lane & 15u)) * CH + 16 * hi;
  const unsigned short* b1 = b0 + (size_t)16 * CH;
#pragma unroll 2
  for (int kk = 0; kk < CH; kk += 32) {
    v16bf a0 = fragA_bf(Apan, CH, kk, lane);
    v16bf a1 = fragA_bf(Apan + 16 * CH, CH, kk, lane);
    v16bf bb0 = frag_from_2(b0 + kk, b0 + kk + 8);
    v16bf bb1 = frag_from_2(b1 + kk, b1 + kk + 8);
    acc[0] = WMMA_BF16(a0, bb0, acc[0]);
    acc[1] = WMMA_BF16(a0, bb1, acc[1]);
    acc[2] = WMMA_BF16(a1, bb0, acc[2]);
    acc[3] = WMMA_BF16(a1, bb1, acc[3]);
  }
}

// Kernel 1: QKV GEMM (M=8192, N=3456 pad 3584, K=1152) -> scatter q/k/vT bf16
__global__ __launch_bounds__(256)
void qkv_gemm_kernel(const unsigned short* __restrict__ xbf,     // [8192][1152]
                     const unsigned short* __restrict__ wT,      // [3584][1152]
                     const float* __restrict__ bias,             // [3456]
                     unsigned short* __restrict__ qbf,           // [BH][1024][96]
                     unsigned short* __restrict__ kbf,           // [BH][1024][96]
                     unsigned short* __restrict__ vT) {          // [BH][96][1024]
  extern __shared__ char smem[];
  unsigned short* Apan = (unsigned short*)smem;                  // 32*1152 bf16 = 72 KB
  const int tid = threadIdx.x;
  const unsigned lane = tid & 31u;
  const int w = tid >> 5;
  const unsigned hi = lane >> 4;
  const int m0 = blockIdx.y * 32;
  const int colw = blockIdx.x * 256 + w * 32;

  copy_panel_to_lds(Apan, xbf + (size_t)m0 * CH, 32 * CH / 8, tid, 256);
  __syncthreads();

  v8f acc[4] = {{}, {}, {}, {}};
  gemm_core_32x32(Apan, wT, colw, lane, acc);

#pragma unroll
  for (int mi = 0; mi < 2; ++mi) {
#pragma unroll
    for (int ni = 0; ni < 2; ++ni) {
      int col = colw + ni * 16 + (int)(lane & 15u);
      if (col < NQKV) {
        float bv = bias[col];
        int which = col / CH;              // 0=q 1=k 2=v
        int rem = col - which * CH;
        int h = rem / HEAD_DIM;
        int d = rem - h * HEAD_DIM;
#pragma unroll
        for (int r = 0; r < 8; ++r) {
          int m = m0 + mi * 16 + r + 8 * (int)hi;
          int b_ = m >> 10;
          int n = m & 1023;
          int bh = b_ * NUM_HEADS + h;
          unsigned short val = f2bf(acc[mi * 2 + ni][r] + bv);
          if (which == 0)      qbf[((size_t)bh * SEQ + n) * HD_PAD + d] = val;
          else if (which == 1) kbf[((size_t)bh * SEQ + n) * HD_PAD + d] = val;
          else                 vT[((size_t)bh * HD_PAD + d) * SEQ + n] = val;
        }
      }
    }
  }
}

// Kernel 3: projection GEMM (M=8192, N=1152 pad 1280, K=1152) -> fp32 d_out
__global__ __launch_bounds__(256)
void proj_gemm_kernel(const unsigned short* __restrict__ abf,    // [8192][1152]
                      const unsigned short* __restrict__ wT,     // [1280][1152]
                      const float* __restrict__ bias,            // [1152]
                      float* __restrict__ out) {                 // [8192][1152]
  extern __shared__ char smem[];
  unsigned short* Apan = (unsigned short*)smem;
  const int tid = threadIdx.x;
  const unsigned lane = tid & 31u;
  const int w = tid >> 5;
  const unsigned hi = lane >> 4;
  const int m0 = blockIdx.y * 32;
  const int colw = blockIdx.x * 256 + w * 32;

  copy_panel_to_lds(Apan, abf + (size_t)m0 * CH, 32 * CH / 8, tid, 256);
  __syncthreads();

  v8f acc[4] = {{}, {}, {}, {}};
  gemm_core_32x32(Apan, wT, colw, lane, acc);

#pragma unroll
  for (int mi = 0; mi < 2; ++mi) {
#pragma unroll
    for (int ni = 0; ni < 2; ++ni) {
      int col = colw + ni * 16 + (int)(lane & 15u);
      if (col < CH) {
        float bv = bias[col];
#pragma unroll
        for (int r = 0; r < 8; ++r) {
          int m = m0 + mi * 16 + r + 8 * (int)hi;
          out[(size_t)m * CH + col] = acc[mi * 2 + ni][r] + bv;
        }
      }
    }
  }
}

// ==================================================================================
// Kernel 2: fused dual attention + losses. grid = BH*64 blocks, 256 threads.
// ==================================================================================
__device__ __forceinline__ void logits_strip(const unsigned short* Qs,      // LDS [16][96]
                                             const unsigned short* kbase,   // k[bh] [1024][96]
                                             float* Ls,                     // LDS [16][1024]
                                             int w, unsigned lane) {
  const unsigned hi = lane >> 4;
  // hoist the 3 Q A-fragments (K = 96 = 3 x 32) into registers
  v16bf a0 = fragA_bf(Qs, HD_PAD, 0, lane);
  v16bf a1 = fragA_bf(Qs, HD_PAD, 32, lane);
  v16bf a2 = fragA_bf(Qs, HD_PAD, 64, lane);
  for (int t = 0; t < 8; ++t) {
    const int m0 = (w * 8 + t) * 16;
    const unsigned short* kb =
        kbase + (size_t)(m0 + (int)(lane & 15u)) * HD_PAD + 16 * hi;
    v8f acc = {};
    acc = WMMA_BF16(a0, frag_from_2(kb, kb + 8), acc);
    acc = WMMA_BF16(a1, frag_from_2(kb + 32, kb + 40), acc);
    acc = WMMA_BF16(a2, frag_from_2(kb + 64, kb + 72), acc);
#pragma unroll
    for (int r = 0; r < 8; ++r)
      Ls[(r + 8 * (int)hi) * 1024 + m0 + (int)(lane & 15u)] = acc[r] * SCALE_F;
  }
}

__global__ __launch_bounds__(256)
void attn_fused_kernel(const unsigned short* __restrict__ qbf,
                       const unsigned short* __restrict__ ksit,
                       const unsigned short* __restrict__ vTsit,
                       const unsigned short* __restrict__ kdino,
                       const unsigned short* __restrict__ vTdino,
                       unsigned short* __restrict__ attn_cat,    // [8192][1152] bf16
                       float* __restrict__ gsum) {               // [2] loss accumulators
  extern __shared__ char smem[];
  float*          Ls  = (float*)smem;                            // 16*1024 f32  = 65536 B
  unsigned short* Ldb = (unsigned short*)(smem + 65536);         // 16*1024 bf16 = 32768 B
  unsigned short* Qs  = (unsigned short*)(smem + 98304);         // 16*96  bf16  =  3072 B
  float*          Asit = (float*)(smem + 101376);                // 16*80 f32    =  5120 B
  float*          Adin = (float*)(smem + 106496);                // 16*80 f32    =  5120 B
  float*          sN  = (float*)(smem + 111616);                 // 16 dino norms
  float*          sM  = sN + 16;                                 // 16 dino rowmax
  float*          sE  = sM + 16;                                 // 16 dino expsum
  float*          bl  = sE + 16;                                 // 2 block loss sums

  const int tid = threadIdx.x;
  const unsigned lane = tid & 31u;
  const int w = tid >> 5;
  const unsigned hi = lane >> 4;
  const int bh = blockIdx.x >> 6;
  const int n0 = (blockIdx.x & 63) * 16;
  const int b = bh >> 4;
  const int h = bh & 15;

  if (tid == 0) { bl[0] = 0.f; bl[1] = 0.f; }
  copy_panel_to_lds(Qs, qbf + ((size_t)bh * SEQ + n0) * HD_PAD, 16 * HD_PAD / 8, tid, 256);
  __syncthreads();

  // ---- dino logits into Ls ----
  logits_strip(Qs, kdino + (size_t)bh * SEQ * HD_PAD, Ls, w, lane);
  __syncthreads();

  // ---- dino row stats; raw logits saved as bf16 ----
  for (int r = 2 * w; r < 2 * w + 2; ++r) {
    float ss = 0.f, mx = -1e30f;
    for (int c = (int)lane; c < 1024; c += 32) {
      float v = Ls[r * 1024 + c];
      Ldb[r * 1024 + c] = f2bf(v);
      ss += v * v;
      mx = fmaxf(mx, v);
    }
    for (int o = 16; o; o >>= 1) {
      ss += __shfl_xor(ss, o, 32);
      mx = fmaxf(mx, __shfl_xor(mx, o, 32));
    }
    float es = 0.f;
    for (int c = (int)lane; c < 1024; c += 32) es += __expf(Ls[r * 1024 + c] - mx);
    for (int o = 16; o; o >>= 1) es += __shfl_xor(es, o, 32);
    if (lane == 0) { sN[r] = fmaxf(sqrtf(ss), EPS_F); sM[r] = mx; sE[r] = es; }
  }
  __syncthreads();

  // ---- sit logits overwrite Ls ----
  logits_strip(Qs, ksit + (size_t)bh * SEQ * HD_PAD, Ls, w, lane);
  __syncthreads();

  // ---- sit stats, cosine(logits) loss, softmax in place ----
  for (int r = 2 * w; r < 2 * w + 2; ++r) {
    float ss = 0.f, mx = -1e30f, dt = 0.f;
    for (int c = (int)lane; c < 1024; c += 32) {
      float v = Ls[r * 1024 + c];
      float dn = bf2f(Ldb[r * 1024 + c]);
      ss += v * v;
      dt += v * dn;
      mx = fmaxf(mx, v);
    }
    for (int o = 16; o; o >>= 1) {
      ss += __shfl_xor(ss, o, 32);
      dt += __shfl_xor(dt, o, 32);
      mx = fmaxf(mx, __shfl_xor(mx, o, 32));
    }
    float es = 0.f;
    for (int c = (int)lane; c < 1024; c += 32) {
      float e = __expf(Ls[r * 1024 + c] - mx);
      Ls[r * 1024 + c] = e;
      es += e;
    }
    for (int o = 16; o; o >>= 1) es += __shfl_xor(es, o, 32);
    float inv = 1.f / es;
    for (int c = (int)lane; c < 1024; c += 32) Ls[r * 1024 + c] *= inv;
    if (lane == 0) {
      float ns = fmaxf(sqrtf(ss), EPS_F);
      atomicAdd(&bl[0], dt / (ns * sN[r]));
    }
  }
  __syncthreads();

  // ---- P @ V: waves 0-3 sit (d-tiles 0..4), waves 4-7 dino (d-tiles 0..4).
  //      kk-outer so the A-fragment conversion is built once per kk per wave. ----
  if (w < 4) {
    const unsigned short* vb = vTsit + (size_t)bh * HD_PAD * SEQ;
    const unsigned short* bp0 =
        vb + ((size_t)(w * 16 + (int)(lane & 15u))) * SEQ + 16 * hi;
    const unsigned short* bp1 =
        vb + ((size_t)(64 + (int)(lane & 15u))) * SEQ + 16 * hi;
    v8f acc0 = {}, acc1 = {};
    for (int kk = 0; kk < 1024; kk += 32) {
      v16bf a = fragA_f32(Ls, 1024, kk, lane);
      acc0 = WMMA_BF16(a, frag_from_2(bp0 + kk, bp0 + kk + 8), acc0);
      if (w == 0)
        acc1 = WMMA_BF16(a, frag_from_2(bp1 + kk, bp1 + kk + 8), acc1);
    }
#pragma unroll
    for (int r = 0; r < 8; ++r)
      Asit[(r + 8 * (int)hi) * 80 + w * 16 + (int)(lane & 15u)] = acc0[r];
    if (w == 0) {
#pragma unroll
      for (int r = 0; r < 8; ++r)
        Asit[(r + 8 * (int)hi) * 80 + 64 + (int)(lane & 15u)] = acc1[r];
    }
  } else {
    const int w4 = w - 4;
    const float mxl = sM[lane & 15u];
    const float invl = 1.f / sE[lane & 15u];
    const unsigned short* vb = vTdino + (size_t)bh * HD_PAD * SEQ;
    const unsigned short* bp0 =
        vb + ((size_t)(w4 * 16 + (int)(lane & 15u))) * SEQ + 16 * hi;
    const unsigned short* bp1 =
        vb + ((size_t)(64 + (int)(lane & 15u))) * SEQ + 16 * hi;
    v8f acc0 = {}, acc1 = {};
    for (int kk = 0; kk < 1024; kk += 32) {
      v16bf a = fragA_dinoP(Ldb, 1024, kk, lane, mxl, invl);
      acc0 = WMMA_BF16(a, frag_from_2(bp0 + kk, bp0 + kk + 8), acc0);
      if (w4 == 0)
        acc1 = WMMA_BF16(a, frag_from_2(bp1 + kk, bp1 + kk + 8), acc1);
    }
#pragma unroll
    for (int r = 0; r < 8; ++r)
      Adin[(r + 8 * (int)hi) * 80 + w4 * 16 + (int)(lane & 15u)] = acc0[r];
    if (w4 == 0) {
#pragma unroll
      for (int r = 0; r < 8; ++r)
        Adin[(r + 8 * (int)hi) * 80 + 64 + (int)(lane & 15u)] = acc1[r];
    }
  }
  __syncthreads();

  // ---- attn cosine loss + write attn_sit for proj GEMM ----
  for (int r = 2 * w; r < 2 * w + 2; ++r) {
    float ss_s = 0.f, ss_d = 0.f, dt = 0.f;
    for (int d = (int)lane; d < HEAD_DIM; d += 32) {
      float a = Asit[r * 80 + d];
      float c = Adin[r * 80 + d];
      ss_s += a * a;
      ss_d += c * c;
      dt += a * c;
    }
    for (int o = 16; o; o >>= 1) {
      ss_s += __shfl_xor(ss_s, o, 32);
      ss_d += __shfl_xor(ss_d, o, 32);
      dt += __shfl_xor(dt, o, 32);
    }
    if (lane == 0) {
      float ns = fmaxf(sqrtf(ss_s), EPS_F);
      float nd = fmaxf(sqrtf(ss_d), EPS_F);
      atomicAdd(&bl[1], dt / (ns * nd));
    }
    for (int d = (int)lane; d < HEAD_DIM; d += 32)
      attn_cat[((size_t)(b * SEQ + n0 + r)) * CH + h * HEAD_DIM + d] = f2bf(Asit[r * 80 + d]);
  }
  __syncthreads();
  if (tid == 0) {
    atomicAdd(&gsum[0], bl[0]);
    atomicAdd(&gsum[1], bl[1]);
  }
}

__global__ void finalize_kernel(const float* __restrict__ gsum, float* __restrict__ loss_out) {
  const float denom = (float)(BATCH * NUM_HEADS * SEQ);
  float l_logit = 1.f - gsum[0] / denom;
  float l_attn = 1.f - gsum[1] / denom;
  loss_out[0] = l_logit + 0.5f * l_attn;
}

// ==================================================================================
// Host launcher
// ==================================================================================
extern "C" void kernel_launch(void* const* d_in, const int* in_sizes, int n_in,
                              void* d_out, int out_size, void* d_ws, size_t ws_size,
                              hipStream_t stream) {
  (void)in_sizes; (void)n_in; (void)out_size; (void)ws_size;
  const float* x      = (const float*)d_in[0];  // [8,1024,1152]
  const float* k_dino = (const float*)d_in[1];  // [8,16,1024,72]
  const float* v_dino = (const float*)d_in[2];
  const float* W_qkv  = (const float*)d_in[3];  // [1152,3456]
  const float* b_qkv  = (const float*)d_in[4];
  const float* W_proj = (const float*)d_in[5];  // [1152,1152]
  const float* b_proj = (const float*)d_in[6];
  float* out = (float*)d_out;                   // 8*1024*1152 floats + 1 loss

  char* ws = (char*)d_ws;
  size_t off = 0;
  auto carve = [&](size_t bytes) -> char* {
    char* p = ws + off;
    off = (off + bytes + 255) & ~(size_t)255;
    return p;
  };
  unsigned short* xbf    = (unsigned short*)carve((size_t)8192 * CH * 2);           // 18.9 MB
  unsigned short* wqkvT  = (unsigned short*)carve((size_t)NQKV_PAD * CH * 2);       //  8.3 MB
  unsigned short* wprojT = (unsigned short*)carve((size_t)NPROJ_PAD * CH * 2);      //  2.9 MB
  unsigned short* qbf    = (unsigned short*)carve((size_t)BHN * SEQ * HD_PAD * 2);  // 25.2 MB
  unsigned short* ksit   = (unsigned short*)carve((size_t)BHN * SEQ * HD_PAD * 2);
  unsigned short* vTsit  = (unsigned short*)carve((size_t)BHN * HD_PAD * SEQ * 2);
  unsigned short* kdino  = (unsigned short*)carve((size_t)BHN * SEQ * HD_PAD * 2);
  unsigned short* vTdino = (unsigned short*)carve((size_t)BHN * HD_PAD * SEQ * 2);
  unsigned short* acat   = (unsigned short*)carve((size_t)8192 * CH * 2);           // 18.9 MB
  float* gsum            = (float*)carve(2 * sizeof(float));

  const int TB = 256;
  // --- init / pack ---
  {  // zero q/k padded buffers (pads must be 0 for the K-loops) + loss accumulators
    long long nz = (long long)BHN * SEQ * HD_PAD / 2;  // uints per buffer
    zero_u32_kernel<<<(unsigned)((nz + TB - 1) / TB), TB, 0, stream>>>((unsigned int*)qbf, nz);
    zero_u32_kernel<<<(unsigned)((nz + TB - 1) / TB), TB, 0, stream>>>((unsigned int*)ksit, nz);
    zero_u32_kernel<<<1, 32, 0, stream>>>((unsigned int*)gsum, 2);
  }
  {
    int n = 8192 * CH;
    f2bf_kernel<<<(n + TB - 1) / TB, TB, 0, stream>>>(x, xbf, n);
  }
  {
    int n = NQKV_PAD * CH;
    transpose_pad_bf_kernel<<<(n + TB - 1) / TB, TB, 0, stream>>>(W_qkv, wqkvT, CH, NQKV, NQKV_PAD);
  }
  {
    int n = NPROJ_PAD * CH;
    transpose_pad_bf_kernel<<<(n + TB - 1) / TB, TB, 0, stream>>>(W_proj, wprojT, CH, CH, NPROJ_PAD);
  }
  {
    int n = BHN * SEQ * HD_PAD;
    pad_bf_kernel<<<(n + TB - 1) / TB, TB, 0, stream>>>(k_dino, kdino, BHN * SEQ);
    padT_bf_kernel<<<(n + TB - 1) / TB, TB, 0, stream>>>(v_dino, vTdino);
  }

  const size_t gemm_smem = (size_t)32 * CH * 2;  // 72 KB A panel
  // --- QKV GEMM ---
  {
    dim3 grid(NQKV_PAD / 256, 8192 / 32);
    qkv_gemm_kernel<<<grid, TB, gemm_smem, stream>>>(xbf, wqkvT, b_qkv, qbf, ksit, vTsit);
  }
  // --- fused dual attention + losses ---
  {
    const size_t smem = 111872;
    attn_fused_kernel<<<BHN * 64, TB, smem, stream>>>(qbf, ksit, vTsit, kdino, vTdino, acat, gsum);
  }
  // --- projection GEMM -> d_out ---
  {
    dim3 grid(NPROJ_PAD / 256, 8192 / 32);
    proj_gemm_kernel<<<grid, TB, gemm_smem, stream>>>(acat, wprojT, b_proj, out);
  }
  // --- loss ---
  finalize_kernel<<<1, 1, 0, stream>>>(gsum, out + (size_t)8192 * CH);
}
